// GSASR_43705587204660
// MI455X (gfx1250) — compile-verified, matrix operand order checked by reading, not run on hardware
//
#include <hip/hip_runtime.h>
#include <hip/hip_bf16.h>
#include <math.h>

// ---------------------------------------------------------------------------
// GSASR forward for MI455X (gfx1250, wave32).  All GEMM-shaped work goes
// through V_WMMA_F32_16X16X32_BF16.  A is fp32 (converted to bf16 with
// vectorized loads); B operands are pre-converted into bf16 [n][k]
// (k-contiguous) layout so every tile load is one 128-bit global load +
// one 128-bit LDS store.  Each wave computes a 32x16 register tile
// (2 WMMAs sharing one B fragment); LDS tiles are double-buffered so the
// K-loop has a single barrier per step.  M must be a multiple of 64 and
// N, K multiples of 32 -- true for every GEMM in this model.
// ---------------------------------------------------------------------------

typedef __attribute__((ext_vector_type(16))) __bf16 v16bf;
typedef __attribute__((ext_vector_type(8)))  __bf16 v8bf;
typedef __attribute__((ext_vector_type(8)))  float  v8f;

#define F_DIM   128
#define N_WIN   16
#define TQ      256
#define N_HEADS 4
#define DH      32
#define HID_DIM 512

__device__ __forceinline__ float geluf(float x) {
  const float kA = 0.7978845608028654f; // sqrt(2/pi)
  return 0.5f * x * (1.0f + tanhf(kA * (x + 0.044715f * x * x * x)));
}
__device__ __forceinline__ float softplusf(float x) {
  return (x > 20.f) ? x : log1pf(__expf(x));
}
__device__ __forceinline__ float waveSum(float v) {
#pragma unroll
  for (int off = 16; off; off >>= 1) v += __shfl_xor(v, off, 32);
  return v;
}
__device__ __forceinline__ float waveMax(float v) {
#pragma unroll
  for (int off = 16; off; off >>= 1) v = fmaxf(v, __shfl_xor(v, off, 32));
  return v;
}

// ---------------------------------------------------------------------------
// Batched GEMM:  C[z] = op(A[z] @ B[z] + bias) (+ addsrc)
//   A: fp32 row-major [m][k] (k-contiguous), B: bf16 [n][k] (k-contiguous)
// Block = 128 threads = 4 waves; macro tile 64(M) x 32(N); wave tile 32x16
// (two 16x16 WMMAs sharing the B fragment).  K stepped by 32 with LDS
// double buffering (one barrier per step).
// Batch z decomposes as (w = z / nH, h = z % nH) with independent strides.
// ---------------------------------------------------------------------------
__global__ __launch_bounds__(128)
void gemm_bf16_wmma(const float* __restrict__ A, const __bf16* __restrict__ B,
                    const float* __restrict__ bias, const float* __restrict__ addsrc,
                    float* __restrict__ C,
                    int M, int N, int K, int lda, int ldb, int ldc,
                    int nH, int sAw, int sAh, int sBw, int sBh, int sCw, int sCh,
                    int doGelu)
{
  __shared__ __align__(16) __bf16 As[2][64][40];   // [buf][m][k], 80B row pitch
  __shared__ __align__(16) __bf16 Bs[2][32][40];   // [buf][n][k]

  const int tid = threadIdx.x;
  const int z   = blockIdx.z;
  const int wz  = z / nH, hz = z % nH;
  const float*  Ab = A + (long)wz * sAw + (long)hz * sAh;
  const __bf16* Bb = B + (long)wz * sBw + (long)hz * sBh;
  float*        Cb = C + (long)wz * sCw + (long)hz * sCh;
  const float*  Rb = addsrc ? (addsrc + (long)wz * sCw + (long)hz * sCh) : nullptr;

  const int mbase = blockIdx.x * 64;
  const int nbase = blockIdx.y * 32;

  const int wv    = tid >> 5;           // wave id 0..3
  const int lane  = tid & 31;
  const int wmB   = (wv >> 1) << 5;     // wave M offset (0/32)
  const int wnB   = (wv & 1) << 4;      // wave N offset (0/16)
  const int mr0   = wmB + (lane & 15);  // A fragment row, tile 0
  const int mr1   = mr0 + 16;           // A fragment row, tile 1
  const int nc    = wnB + (lane & 15);  // B fragment col
  const int khalf = (lane >> 4) << 3;   // K half select per ISA layout

  // per-thread tile-load coordinates: 8 contiguous k per thread
  const int lr = tid >> 2;              // 0..31
  const int ks = (tid & 3) << 3;        // 0,8,16,24

  const float*  aSrc0 = Ab + (long)(mbase + lr) * lda + ks;
  const float*  aSrc1 = aSrc0 + (long)32 * lda;
  const __bf16* bSrc  = Bb + (long)(nbase + lr) * ldb + ks;

  auto loadTiles = [&](int k0, int p) {
    {
      const float4* s4 = (const float4*)(aSrc0 + k0);
      float4 f0 = s4[0], f1 = s4[1];
      v8bf h;
      h[0] = (__bf16)f0.x; h[1] = (__bf16)f0.y; h[2] = (__bf16)f0.z; h[3] = (__bf16)f0.w;
      h[4] = (__bf16)f1.x; h[5] = (__bf16)f1.y; h[6] = (__bf16)f1.z; h[7] = (__bf16)f1.w;
      *(v8bf*)&As[p][lr][ks] = h;
    }
    {
      const float4* s4 = (const float4*)(aSrc1 + k0);
      float4 f0 = s4[0], f1 = s4[1];
      v8bf h;
      h[0] = (__bf16)f0.x; h[1] = (__bf16)f0.y; h[2] = (__bf16)f0.z; h[3] = (__bf16)f0.w;
      h[4] = (__bf16)f1.x; h[5] = (__bf16)f1.y; h[6] = (__bf16)f1.z; h[7] = (__bf16)f1.w;
      *(v8bf*)&As[p][lr + 32][ks] = h;
    }
    {
      v8bf b8 = *(const v8bf*)(bSrc + k0);
      *(v8bf*)&Bs[p][lr][ks] = b8;
    }
  };

  v8f acc0 = {}, acc1 = {};

  loadTiles(0, 0);
  __syncthreads();

  int p = 0;
  for (int k0 = 0; k0 < K; k0 += 32) {
    if (k0 + 32 < K) loadTiles(k0 + 32, p ^ 1);
    if (k0 + 64 < K) {
      __builtin_prefetch(aSrc0 + k0 + 64, 0, 1);
      __builtin_prefetch(aSrc1 + k0 + 64, 0, 1);
      __builtin_prefetch(bSrc  + k0 + 64, 0, 1);
    }

    // fragments per CDNA5 16-bit A/B VGPR layout (merge to ds_load_b128)
    v16bf a0, a1, bv;
#pragma unroll
    for (int j = 0; j < 8; ++j) {
      int kk = ((j >> 2) << 4) + khalf + ((j & 3) << 1);
      a0[2 * j]     = As[p][mr0][kk];
      a0[2 * j + 1] = As[p][mr0][kk + 1];
      a1[2 * j]     = As[p][mr1][kk];
      a1[2 * j + 1] = As[p][mr1][kk + 1];
      bv[2 * j]     = Bs[p][nc][kk];
      bv[2 * j + 1] = Bs[p][nc][kk + 1];
    }
    acc0 = __builtin_amdgcn_wmma_f32_16x16x32_bf16(false, a0, false, bv,
                                                   (short)0, acc0, false, false);
    acc1 = __builtin_amdgcn_wmma_f32_16x16x32_bf16(false, a1, false, bv,
                                                   (short)0, acc1, false, false);
    __syncthreads();
    p ^= 1;
  }

  // epilogue: C/D layout VGPR v -> row = v + 8*(lane>=16), col = lane&15
  const int col  = nbase + nc;
  const int rhi  = (lane >> 4) << 3;
  float biasv = bias ? bias[col] : 0.0f;
#pragma unroll
  for (int v = 0; v < 8; ++v) {
    int row = mbase + wmB + v + rhi;
    float val = acc0[v] + biasv;
    if (doGelu) val = geluf(val);
    if (Rb)     val += Rb[(long)row * ldc + col];
    Cb[(long)row * ldc + col] = val;
  }
#pragma unroll
  for (int v = 0; v < 8; ++v) {
    int row = mbase + wmB + 16 + v + rhi;
    float val = acc1[v] + biasv;
    if (doGelu) val = geluf(val);
    if (Rb)     val += Rb[(long)row * ldc + col];
    Cb[(long)row * ldc + col] = val;
  }
}

// ---------------------------------------------------------------------------
// bf16 conversion kernels (prepare B operands in [n][k] layout)
// ---------------------------------------------------------------------------
__global__ void cvt_bf16(const float* __restrict__ in, __bf16* __restrict__ out, int n)
{
  int i = blockIdx.x * blockDim.x + threadIdx.x;
  if (i < n) out[i] = (__bf16)in[i];
}

// W fp32 [K][N] -> Wt bf16 [N][K]
__global__ void cvt_bf16_T(const float* __restrict__ in, __bf16* __restrict__ out,
                           int K, int N)
{
  int i = blockIdx.x * blockDim.x + threadIdx.x;
  if (i >= K * N) return;
  int k = i % K, n = i / K;
  out[i] = (__bf16)in[(long)k * N + n];
}

// V fp32 [w*Tk+k][128] -> Vt bf16 [(w*4+h)*32+n][Tk]   (per-head transpose)
__global__ void cvt_vt(const float* __restrict__ V, __bf16* __restrict__ Vt, int Tk)
{
  int i = blockIdx.x * blockDim.x + threadIdx.x;
  int total = N_WIN * N_HEADS * DH * Tk;
  if (i >= total) return;
  int k = i % Tk, t = i / Tk;
  int n = t & 31, wh = t >> 5;
  int h = wh & 3, w = wh >> 2;
  Vt[i] = (__bf16)V[((long)w * Tk + k) * 128 + h * 32 + n];
}

// ---------------------------------------------------------------------------
// LayerNorm over last dim (=128), one wave per row.
// ---------------------------------------------------------------------------
__global__ __launch_bounds__(128)
void ln128(const float* __restrict__ in, float* __restrict__ out, int nrows)
{
  int wv = threadIdx.x >> 5, lane = threadIdx.x & 31;
  int row = blockIdx.x * 4 + wv;
  if (row >= nrows) return;
  const float* p = in + (long)row * 128;
  float v[4], s = 0.f, s2 = 0.f;
#pragma unroll
  for (int i = 0; i < 4; ++i) { v[i] = p[lane + 32 * i]; s += v[i]; s2 += v[i] * v[i]; }
  s = waveSum(s); s2 = waveSum(s2);
  float mu  = s * (1.f / 128.f);
  float var = s2 * (1.f / 128.f) - mu * mu;
  float inv = rsqrtf(var + 1e-5f);
  float* q = out + (long)row * 128;
#pragma unroll
  for (int i = 0; i < 4; ++i) q[lane + 32 * i] = (v[i] - mu) * inv;
}

// ---------------------------------------------------------------------------
// Row softmax (row length L in {64,256}), in place, one wave per row.
// ---------------------------------------------------------------------------
__global__ __launch_bounds__(128)
void softmax_rows(float* __restrict__ s, int nrows, int L, float scale)
{
  int wv = threadIdx.x >> 5, lane = threadIdx.x & 31;
  int row = blockIdx.x * 4 + wv;
  if (row >= nrows) return;
  float* p = s + (long)row * L;
  int per = L >> 5;            // 2 or 8
  float v[8], m = -1e30f;
  for (int i = 0; i < per; ++i) { v[i] = p[lane + 32 * i] * scale; m = fmaxf(m, v[i]); }
  m = waveMax(m);
  float sum = 0.f;
  for (int i = 0; i < per; ++i) { v[i] = __expf(v[i] - m); sum += v[i]; }
  sum = waveSum(sum);
  float r = 1.f / sum;
  for (int i = 0; i < per; ++i) p[lane + 32 * i] = v[i] * r;
}

// ---------------------------------------------------------------------------
// 3x3 SAME conv (3->128) directly into windowed layout (16 windows, 64 tok).
// ---------------------------------------------------------------------------
__global__ void conv_win_kernel(const float* __restrict__ img, const float* __restrict__ wgt,
                                const float* __restrict__ b, float* __restrict__ lrw)
{
  int idx = blockIdx.x * blockDim.x + threadIdx.x;
  if (idx >= 32 * 32 * 128) return;
  int f = idx & 127, pos = idx >> 7;
  int y = pos >> 5, x = pos & 31;
  float acc = b[f];
#pragma unroll
  for (int c = 0; c < 3; ++c)
#pragma unroll
    for (int ky = 0; ky < 3; ++ky) {
      int yy = y + ky - 1; if (yy < 0 || yy > 31) continue;
#pragma unroll
      for (int kx = 0; kx < 3; ++kx) {
        int xx = x + kx - 1; if (xx < 0 || xx > 31) continue;
        acc += img[(c * 32 + yy) * 32 + xx] * wgt[((f * 3 + c) * 3 + ky) * 3 + kx];
      }
    }
  int hw = y >> 3, r = y & 7, ww = x >> 3, cc = x & 7;
  lrw[(((hw * 4 + ww) * 64) + (r * 8 + cc)) * 128 + f] = acc;
}

__global__ void bcast_emb(const float* __restrict__ emb, float* __restrict__ X, int total)
{
  int i = blockIdx.x * blockDim.x + threadIdx.x;
  if (i < total) X[i] = emb[i & (32768 - 1)];
}

__global__ void add_bcast128(float* __restrict__ X, const float* __restrict__ sf, int total)
{
  int i = blockIdx.x * blockDim.x + threadIdx.x;
  if (i < total) X[i] += sf[i & 127];
}

// scale_feat = gelu(s @ w1 + b1) @ w2 + b2  (s is a scalar) -> 128 values
__global__ void scale_mlp(const int* __restrict__ sf_int,
                          const float* __restrict__ w1, const float* __restrict__ b1,
                          const float* __restrict__ w2, const float* __restrict__ b2,
                          float* __restrict__ outf)
{
  int f = threadIdx.x;                        // 128 threads
  float s = (float)(*sf_int);
  float acc = b2[f];
  for (int j = 0; j < 512; ++j) {
    float h = geluf(s * w1[j] + b1[j]);
    acc += h * w2[j * 128 + f];
  }
  outf[f] = acc;
}

// ---------------------------------------------------------------------------
// Gaussian head: 9 projections per token + nonlinearities; handles the
// window->global spatial rearrange in the index math.
// ---------------------------------------------------------------------------
__global__ void head_kernel(const float* __restrict__ X,
                            const float* opw, const float* opb,
                            const float* rhow, const float* rhob,
                            const float* mw, const float* mb,
                            const float* sw, const float* sb,
                            const float* cw, const float* cb,
                            float* __restrict__ gop, float* __restrict__ grho,
                            float* __restrict__ gme, float* __restrict__ gsd,
                            float* __restrict__ gcl)
{
  int n = blockIdx.x * blockDim.x + threadIdx.x;
  if (n >= 4096) return;
  int gr = n >> 6, gc = n & 63;
  int hw = gr >> 4, r = gr & 15, ww = gc >> 4, c = gc & 15;
  const float* xv = X + (long)(((hw * 4 + ww) * 256) + (r * 16 + c)) * 128;
  float d_op = opb[0], d_rho = rhob[0];
  float d_mx = mb[0], d_my = mb[1], d_sx = sb[0], d_sy = sb[1];
  float d_c0 = cb[0], d_c1 = cb[1], d_c2 = cb[2];
  for (int f = 0; f < 128; ++f) {
    float xf = xv[f];
    d_op  += xf * opw[f];     d_rho += xf * rhow[f];
    d_mx  += xf * mw[f * 2];  d_my  += xf * mw[f * 2 + 1];
    d_sx  += xf * sw[f * 2];  d_sy  += xf * sw[f * 2 + 1];
    d_c0  += xf * cw[f * 3];  d_c1  += xf * cw[f * 3 + 1]; d_c2 += xf * cw[f * 3 + 2];
  }
  gop[n]  = 1.f / (1.f + __expf(-d_op));
  grho[n] = 0.99f * tanhf(d_rho);
  float gx = 32.f * (float)gc / 63.f;   // linspace(0, 32, 64)
  float gy = 32.f * (float)gr / 63.f;
  gme[2 * n]     = gx + tanhf(d_mx);
  gme[2 * n + 1] = gy + tanhf(d_my);
  gsd[2 * n]     = softplusf(d_sx) + 0.3f;
  gsd[2 * n + 1] = softplusf(d_sy) + 0.3f;
  gcl[3 * n] = d_c0; gcl[3 * n + 1] = d_c1; gcl[3 * n + 2] = d_c2;
}

// ---------------------------------------------------------------------------
// Rasterize: one thread per pixel over all 4096 gaussians (L2-resident).
// Output layout (3, Hout, Wout).
// ---------------------------------------------------------------------------
__global__ void raster_kernel(const float* __restrict__ gop, const float* __restrict__ grho,
                              const float* __restrict__ gme, const float* __restrict__ gsd,
                              const float* __restrict__ gcl,
                              float* __restrict__ out, int P, int Wout, float inv_sf)
{
  int p = blockIdx.x * blockDim.x + threadIdx.x;
  if (p >= P) return;
  int i = p / Wout, j = p % Wout;
  float px = ((float)j + 0.5f) * inv_sf;
  float py = ((float)i + 0.5f) * inv_sf;
  float a0 = 0.f, a1 = 0.f, a2 = 0.f;
  for (int n = 0; n < 4096; ++n) {
    float dx = px - gme[2 * n], dy = py - gme[2 * n + 1];
    float ax = dx / gsd[2 * n], ay = dy / gsd[2 * n + 1];
    float rr = grho[n];
    float quad = ax * ax - 2.f * rr * ax * ay + ay * ay;
    float g = __expf(-0.5f * quad / (1.f - rr * rr));
    if (g > 0.1f) {
      float w = gop[n] * g;
      a0 += w * gcl[3 * n]; a1 += w * gcl[3 * n + 1]; a2 += w * gcl[3 * n + 2];
    }
  }
  out[0 * P + p] = a0; out[1 * P + p] = a1; out[2 * P + p] = a2;
}

// ---------------------------------------------------------------------------
// Host-side orchestration
// ---------------------------------------------------------------------------
struct BlockP { const float *qw,*qb,*kw,*kb,*vw,*vb,*ow,*ob,*w1,*b1,*w2,*b2; };

static BlockP getBlock(void* const* din, int base) {
  BlockP p;
  p.qw = (const float*)din[base + 0];  p.qb = (const float*)din[base + 1];
  p.kw = (const float*)din[base + 2];  p.kb = (const float*)din[base + 3];
  p.vw = (const float*)din[base + 4];  p.vb = (const float*)din[base + 5];
  p.ow = (const float*)din[base + 6];  p.ob = (const float*)din[base + 7];
  p.w1 = (const float*)din[base + 8];  p.b1 = (const float*)din[base + 9];
  p.w2 = (const float*)din[base + 10]; p.b2 = (const float*)din[base + 11];
  return p;
}

struct Bf16Ws {
  __bf16 *Wq, *Wk, *Wv, *Wo, *W1t, *W2t, *Kbf, *Vt;
};

static inline void g_gemm(hipStream_t st, const float* A, const __bf16* B, const float* bias,
                          const float* addsrc, float* C, int M, int N, int K,
                          int lda, int ldb, int ldc,
                          int batch = 1, int nH = 1,
                          int sAw = 0, int sAh = 0, int sBw = 0, int sBh = 0,
                          int sCw = 0, int sCh = 0, int doGelu = 0)
{
  dim3 grid(M / 64, N / 32, batch), blk(128);
  gemm_bf16_wmma<<<grid, blk, 0, st>>>(A, B, bias, addsrc, C, M, N, K, lda, ldb, ldc,
                                       nH, sAw, sAh, sBw, sBh, sCw, sCh, doGelu);
}

static void run_block(hipStream_t st, float* X, const float* KV, int Tk, const BlockP& bp,
                      float* LN, float* LNKV, float* Q, float* Kb, float* Vb,
                      float* S, float* Hh, const Bf16Ws& bw)
{
  const int Mq = N_WIN * TQ;   // 4096
  const int Mk = N_WIN * Tk;

  // convert this block's weights to bf16 [N][K] once (stream-ordered reuse)
  cvt_bf16_T<<<(128 * 128 + 255) / 256, 256, 0, st>>>(bp.qw, bw.Wq, 128, 128);
  cvt_bf16_T<<<(128 * 128 + 255) / 256, 256, 0, st>>>(bp.kw, bw.Wk, 128, 128);
  cvt_bf16_T<<<(128 * 128 + 255) / 256, 256, 0, st>>>(bp.vw, bw.Wv, 128, 128);
  cvt_bf16_T<<<(128 * 128 + 255) / 256, 256, 0, st>>>(bp.ow, bw.Wo, 128, 128);
  cvt_bf16_T<<<(128 * 512 + 255) / 256, 256, 0, st>>>(bp.w1, bw.W1t, 128, 512);
  cvt_bf16_T<<<(512 * 128 + 255) / 256, 256, 0, st>>>(bp.w2, bw.W2t, 512, 128);

  ln128<<<(Mq + 3) / 4, 128, 0, st>>>(X, LN, Mq);
  const float* lnkv = LN;
  if (KV != (const float*)X) {
    ln128<<<(Mk + 3) / 4, 128, 0, st>>>(KV, LNKV, Mk);
    lnkv = LNKV;
  }
  g_gemm(st, LN,   bw.Wq, bp.qb, nullptr, Q,  Mq, 128, 128, 128, 128, 128);
  g_gemm(st, lnkv, bw.Wk, bp.kb, nullptr, Kb, Mk, 128, 128, 128, 128, 128);
  g_gemm(st, lnkv, bw.Wv, bp.vb, nullptr, Vb, Mk, 128, 128, 128, 128, 128);

  // bf16 copies of K (row-major == [n][k] for Q@K^T) and per-head V^T
  cvt_bf16<<<(Mk * 128 + 255) / 256, 256, 0, st>>>(Kb, bw.Kbf, Mk * 128);
  cvt_vt<<<(N_WIN * N_HEADS * DH * Tk + 255) / 256, 256, 0, st>>>(Vb, bw.Vt, Tk);

  // scores[w,h] = q_h @ k_h^T  (M=256, N=Tk, K=32), batch over 64 (w,h)
  g_gemm(st, Q, bw.Kbf, nullptr, nullptr, S, TQ, Tk, DH, 128, 128, Tk,
         N_WIN * N_HEADS, N_HEADS,
         TQ * 128, DH, Tk * 128, DH,
         N_HEADS * TQ * Tk, TQ * Tk);
  softmax_rows<<<(N_WIN * N_HEADS * TQ + 3) / 4, 128, 0, st>>>(
      S, N_WIN * N_HEADS * TQ, Tk, 0.17677669529663687f /* 1/sqrt(32) */);
  // o[w,:,h*32:h*32+32] = att @ v_h   (reuse LN as o buffer)
  g_gemm(st, S, bw.Vt, nullptr, nullptr, LN, TQ, DH, Tk, Tk, Tk, 128,
         N_WIN * N_HEADS, N_HEADS,
         N_HEADS * TQ * Tk, TQ * Tk,
         N_HEADS * DH * Tk, DH * Tk,
         TQ * 128, DH);
  // X = X + o @ ow + ob
  g_gemm(st, LN, bw.Wo, bp.ob, X, X, Mq, 128, 128, 128, 128, 128);
  // MLP: X = X + gelu(ln(X) @ w1 + b1) @ w2 + b2
  ln128<<<(Mq + 3) / 4, 128, 0, st>>>(X, LN, Mq);
  g_gemm(st, LN, bw.W1t, bp.b1, nullptr, Hh, Mq, 512, 128, 128, 128, 512,
         1, 1, 0, 0, 0, 0, 0, 0, /*gelu*/1);
  g_gemm(st, Hh, bw.W2t, bp.b2, X, X, Mq, 128, 512, 512, 512, 128);
}

extern "C" void kernel_launch(void* const* d_in, const int* in_sizes, int n_in,
                              void* d_out, int out_size, void* d_ws, size_t ws_size,
                              hipStream_t stream)
{
  (void)in_sizes;
  if (n_in < 55) return;   // expects flattened params in dict insertion order

  const float* img   = (const float*)d_in[0];
  const int*   sfac  = (const int*)d_in[1];
  const float* enc_w = (const float*)d_in[2];
  const float* enc_b = (const float*)d_in[3];
  const float* emb   = (const float*)d_in[4];
  BlockP cib  = getBlock(d_in, 5);
  const float* sw1 = (const float*)d_in[17];
  const float* sb1 = (const float*)d_in[18];
  const float* sw2 = (const float*)d_in[19];
  const float* sb2 = (const float*)d_in[20];
  BlockP gib0 = getBlock(d_in, 21);
  BlockP gib1 = getBlock(d_in, 33);
  const float* hp[10];
  for (int i = 0; i < 10; ++i) hp[i] = (const float*)d_in[45 + i];

  // workspace layout (fp32 region, counts in floats)
  float* ws   = (float*)d_ws;
  float* LRW  = ws;                 // 16*64*128
  float* LNKV = LRW  + 131072;      // 16*64*128
  float* X    = LNKV + 131072;      // 16*256*128
  float* LN   = X    + 524288;      // ln buffer / attention 'o'
  float* Q    = LN   + 524288;
  float* Kb   = Q    + 524288;
  float* Vb   = Kb   + 524288;
  float* S    = Vb   + 524288;      // 16*4*256*256
  float* Hh   = S    + 4194304;     // 16*256*512
  float* SF   = Hh   + 2097152;     // 128
  float* GOP  = SF   + 128;
  float* GRHO = GOP  + 4096;
  float* GME  = GRHO + 4096;
  float* GSD  = GME  + 8192;
  float* GCL  = GSD  + 8192;        // + 12288

  // bf16 region (counts in bf16 elements), starts 16B-aligned
  __bf16* bws = (__bf16*)(GCL + 12288);
  Bf16Ws bw;
  bw.Wq  = bws;                     // 128*128
  bw.Wk  = bw.Wq  + 16384;
  bw.Wv  = bw.Wk  + 16384;
  bw.Wo  = bw.Wv  + 16384;
  bw.W1t = bw.Wo  + 16384;          // [512][128]
  bw.W2t = bw.W1t + 65536;          // [128][512]
  bw.Kbf = bw.W2t + 65536;          // up to 4096*128
  bw.Vt  = bw.Kbf + 524288;         // up to 64*32*256
  size_t need = (size_t)((char*)(bw.Vt + 524288) - (char*)d_ws);
  if (ws_size < need) return;

  // 1) conv encoder -> windowed kv
  conv_win_kernel<<<(131072 + 255) / 256, 256, 0, stream>>>(img, enc_w, enc_b, LRW);
  // 2) x = broadcast(embedding)
  bcast_emb<<<(524288 + 255) / 256, 256, 0, stream>>>(emb, X, 524288);
  // 3) cross block (kv = conv features, Tk=64)
  run_block(stream, X, LRW, 64, cib, LN, LNKV, Q, Kb, Vb, S, Hh, bw);
  // 4) scale feature MLP
  scale_mlp<<<1, 128, 0, stream>>>(sfac, sw1, sb1, sw2, sb2, SF);
  // 5) two self blocks (Tk=256), each preceded by +scale_feat broadcast
  add_bcast128<<<(524288 + 255) / 256, 256, 0, stream>>>(X, SF, 524288);
  run_block(stream, X, X, 256, gib0, LN, LNKV, Q, Kb, Vb, S, Hh, bw);
  add_bcast128<<<(524288 + 255) / 256, 256, 0, stream>>>(X, SF, 524288);
  run_block(stream, X, X, 256, gib1, LN, LNKV, Q, Kb, Vb, S, Hh, bw);
  // 6) gaussian head
  head_kernel<<<(4096 + 127) / 128, 128, 0, stream>>>(
      X, hp[0], hp[1], hp[2], hp[3], hp[4], hp[5], hp[6], hp[7], hp[8], hp[9],
      GOP, GRHO, GME, GSD, GCL);
  // 7) rasterize to output (3, Hout, Wout)
  int P = out_size / 3;
  int Wout = (int)(sqrtf((float)P) + 0.5f);
  float inv_sf = 32.0f / (float)Wout;    // pixel coords back in input units
  raster_kernel<<<(P + 127) / 128, 128, 0, stream>>>(
      GOP, GRHO, GME, GSD, GCL, (float*)d_out, P, Wout, inv_sf);
}
